// SGFormer_9706626089235
// MI455X (gfx1250) — matrix operand order repaired and checked by reading, not compiled, not run on hardware
//
#include <hip/hip_runtime.h>
#include <hip/hip_bf16.h>

// ---------------------------------------------------------------------------
// SGFormer-style fused pipeline for MI455X (gfx1250, wave32, WMMA).
// All GEMMs use v_wmma_f32_16x16x32_f16 (f16 in, f32 accumulate).
// ---------------------------------------------------------------------------

typedef __attribute__((ext_vector_type(16))) _Float16 v16h;
typedef __attribute__((ext_vector_type(8)))  _Float16 v8h;
typedef __attribute__((ext_vector_type(4)))  _Float16 v4h;
typedef __attribute__((ext_vector_type(8)))  float    v8f;

#define NN     50000
#define EE     600000
#define DD     128
#define NTILE  3125          // NN / 16
#define NPAD   50016         // NN rounded up to 32 (K padding for k^T v GEMM)
#define NCHUNK 1563          // NPAD / 32
#define KSPLIT 32

static __device__ __forceinline__ v8f wmma_f16(v16h a, v16h b, v8f c) {
  // D = A(16x32 f16) * B(32x16 f16) + C(16x16 f32)
  return __builtin_amdgcn_wmma_f32_16x16x32_f16(
      /*neg_a=*/false, a, /*neg_b=*/false, b,
      /*c_mod=*/(short)0, c, /*reuse_a=*/false, /*reuse_b=*/false);
}

// A-operand layout (16-bit A 16x32, ISA 7.12.2): lane L holds row M=L&15,
// halves 0..7 = K(base..base+7), halves 8..15 = K(base+16..base+23),
// base = (L<16 ? 0 : 8).  -> two contiguous 16B loads at p and p+16 halves.
static __device__ __forceinline__ v16h load_a16(const _Float16* p) {
  union { v16h v; v8h h[2]; } u;
  u.h[0] = *(const v8h*)p;
  u.h[1] = *(const v8h*)(p + 16);
  return u.v;
}

// ---------------------------------------------------------------------------
// K0: pack weights into WMMA B layout (f16), zero accumulators & pads.
// B layout (32x16 16-bit, ISA 7.12.5 style): lane L = column (L&15);
// lanes 0-15 hold K=kc*32+{0..7,16..23}, lanes 16-31 hold K=kc*32+{8..15,24..31}.
// Packed index: ((mat*8 + ct)*4 + kc)*512 + lane*16 + h
// ---------------------------------------------------------------------------
__global__ __launch_bounds__(256) void prep_kernel(
    const float* __restrict__ Wq, const float* __restrict__ Wk,
    const float* __restrict__ Wv, const float* __restrict__ Wg,
    const float* __restrict__ Wl,
    _Float16* __restrict__ wpack, float* __restrict__ Mbuf,
    float* __restrict__ ksum, float* __restrict__ scal,
    _Float16* __restrict__ kT, _Float16* __restrict__ vT)
{
  int gid = blockIdx.x * blockDim.x + threadIdx.x;
  int stride = gridDim.x * blockDim.x;
  // 5 matrices * 16384 packed halves
  for (int i = gid; i < 5 * 16384; i += stride) {
    int mat = i / 16384, r = i % 16384;
    int t = r / 2048;  r %= 2048;
    int c = r / 512;   r %= 512;
    int lane = r / 16; int h = r % 16;
    int n = t * 16 + (lane & 15);
    int k = c * 32 + ((lane < 16) ? 0 : 8) + ((h < 8) ? h : h + 8);
    const float* W = (mat == 0) ? Wq : (mat == 1) ? Wk : (mat == 2) ? Wv
                    : (mat == 3) ? Wg : Wl;
    // Wq/Wk/Wv are [in,out] used as x@W -> B[k][n]=W[k][n].
    // W_gcn / W_lin are [out,in] used as x@W^T -> B[k][n]=W[n][k].
    float v = (mat < 3) ? W[k * DD + n] : W[n * DD + k];
    wpack[i] = (_Float16)v;
  }
  for (int i = gid; i < DD * DD; i += stride) Mbuf[i] = 0.0f;
  for (int i = gid; i < DD;      i += stride) ksum[i] = 0.0f;
  for (int i = gid; i < 8;       i += stride) scal[i] = 0.0f;
  // zero pad columns NN..NPAD of kT / vT (rows = 128 features)
  for (int i = gid; i < DD * (NPAD - NN); i += stride) {
    int row = i / (NPAD - NN), cc = i % (NPAD - NN);
    kT[(size_t)row * NPAD + NN + cc] = (_Float16)0.0f;
    vT[(size_t)row * NPAD + NN + cc] = (_Float16)0.0f;
  }
}

// K1: x (f32) -> xh (f16), row-major [N,128]
__global__ __launch_bounds__(256) void xcast_kernel(
    const float* __restrict__ x, _Float16* __restrict__ xh)
{
  int i = blockIdx.x * blockDim.x + threadIdx.x;
  if (i < NN * DD / 4) {
    float4 f = ((const float4*)x)[i];
    v4h h;
    h[0] = (_Float16)f.x; h[1] = (_Float16)f.y;
    h[2] = (_Float16)f.z; h[3] = (_Float16)f.w;
    ((v4h*)xh)[i] = h;
  }
}

// K2: degrees with self loops
__global__ __launch_bounds__(256) void deg_init_kernel(float* __restrict__ deg) {
  int i = blockIdx.x * blockDim.x + threadIdx.x;
  if (i < NN) deg[i] = 1.0f;  // self loop
}
__global__ __launch_bounds__(256) void deg_edge_kernel(
    const int* __restrict__ dst, float* __restrict__ deg) {
  int i = blockIdx.x * blockDim.x + threadIdx.x;
  if (i < EE) atomicAdd(&deg[dst[i]], 1.0f);
}
__global__ __launch_bounds__(256) void deg_fin_kernel(
    const float* __restrict__ deg, float* __restrict__ dinv,
    float* __restrict__ ideg) {
  int i = blockIdx.x * blockDim.x + threadIdx.x;
  if (i < NN) {
    float d = deg[i];
    dinv[i] = rsqrtf(d);
    ideg[i] = 1.0f / d;
  }
}

// ---------------------------------------------------------------------------
// K3: fused 4-way GEMM.  One wave per 16-row tile of x.
//   q  -> f16 row-major          (A operand of final GEMMs)
//   k  -> kT f16 [128][NPAD]     (+ ksum, ||k||^2)
//   v  -> f32 row-major + vT f16 (+ nothing)
//   xw -> f32 row-major, gcn <- xw/deg (self-loop init)
// ---------------------------------------------------------------------------
__global__ __launch_bounds__(256) void gemm_qkvg_kernel(
    const _Float16* __restrict__ xh, const _Float16* __restrict__ wpack,
    const float* __restrict__ ideg,
    _Float16* __restrict__ q16, float* __restrict__ v32,
    float* __restrict__ xw, float* __restrict__ gcn,
    _Float16* __restrict__ kT, _Float16* __restrict__ vT,
    float* __restrict__ ksum, float* __restrict__ scal)
{
  const int lane = threadIdx.x & 31;
  const int wave = threadIdx.x >> 5;
  const int tile = blockIdx.x * 8 + wave;
  if (tile >= NTILE) return;              // uniform per wave -> EXEC stays full
  const int m0   = tile * 16;
  const int hi   = lane >> 4;
  const int colL = lane & 15;
  const int rowA = m0 + colL;
  const int baseA = hi * 8;

  // A tiles: x rows, 4 chunks of K=32
  v16h A[4];
#pragma unroll
  for (int c = 0; c < 4; ++c)
    A[c] = load_a16(xh + (size_t)rowA * DD + c * 32 + baseA);

  float idg[8];
#pragma unroll
  for (int r = 0; r < 8; ++r) idg[r] = ideg[m0 + r + hi * 8];

  float pq = 0.0f, pk = 0.0f;

  // ---- q = x @ Wq ----
#pragma unroll
  for (int ct = 0; ct < 8; ++ct) {
    v8f acc = {0, 0, 0, 0, 0, 0, 0, 0};
#pragma unroll
    for (int c = 0; c < 4; ++c)
      acc = wmma_f16(A[c], *(const v16h*)(wpack + ((0 * 8 + ct) * 4 + c) * 512 + lane * 16), acc);
    int col = ct * 16 + colL;
#pragma unroll
    for (int r = 0; r < 8; ++r) {
      float vv = acc[r];
      pq += vv * vv;
      q16[(size_t)(m0 + r + hi * 8) * DD + col] = (_Float16)vv;
    }
  }
  // ---- k = x @ Wk ----
#pragma unroll
  for (int ct = 0; ct < 8; ++ct) {
    v8f acc = {0, 0, 0, 0, 0, 0, 0, 0};
#pragma unroll
    for (int c = 0; c < 4; ++c)
      acc = wmma_f16(A[c], *(const v16h*)(wpack + ((1 * 8 + ct) * 4 + c) * 512 + lane * 16), acc);
    int col = ct * 16 + colL;
    float s8 = 0.0f;
    v8h hv;
#pragma unroll
    for (int r = 0; r < 8; ++r) {
      float vv = acc[r];
      pk += vv * vv;
      s8 += vv;
      hv[r] = (_Float16)vv;
    }
    // k^T store: column 'col' of k = row 'col' of kT; 8 consecutive rows
    *(v8h*)(kT + (size_t)col * NPAD + m0 + hi * 8) = hv;
    float s = s8 + __shfl_xor(s8, 16);
    if (lane < 16) atomicAdd(&ksum[col], s);
  }
  // ---- v = x @ Wv ----
#pragma unroll
  for (int ct = 0; ct < 8; ++ct) {
    v8f acc = {0, 0, 0, 0, 0, 0, 0, 0};
#pragma unroll
    for (int c = 0; c < 4; ++c)
      acc = wmma_f16(A[c], *(const v16h*)(wpack + ((2 * 8 + ct) * 4 + c) * 512 + lane * 16), acc);
    int col = ct * 16 + colL;
    v8h hv;
#pragma unroll
    for (int r = 0; r < 8; ++r) {
      float vv = acc[r];
      v32[(size_t)(m0 + r + hi * 8) * DD + col] = vv;
      hv[r] = (_Float16)vv;
    }
    *(v8h*)(vT + (size_t)col * NPAD + m0 + hi * 8) = hv;
  }
  // ---- xw = x @ W_gcn^T ; gcn init = xw / deg (self loop) ----
#pragma unroll
  for (int ct = 0; ct < 8; ++ct) {
    v8f acc = {0, 0, 0, 0, 0, 0, 0, 0};
#pragma unroll
    for (int c = 0; c < 4; ++c)
      acc = wmma_f16(A[c], *(const v16h*)(wpack + ((3 * 8 + ct) * 4 + c) * 512 + lane * 16), acc);
    int col = ct * 16 + colL;
#pragma unroll
    for (int r = 0; r < 8; ++r) {
      float vv = acc[r];
      int row = m0 + r + hi * 8;
      xw[(size_t)row * DD + col]  = vv;
      gcn[(size_t)row * DD + col] = vv * idg[r];
    }
  }

  atomicAdd(&scal[0], pq);
  atomicAdd(&scal[1], pk);
}

// K4: GCN edge scatter. One wave per edge, 4 f32 atomics per lane.
__global__ __launch_bounds__(256) void gcn_scatter_kernel(
    const int* __restrict__ src, const int* __restrict__ dst,
    const float* __restrict__ dinv, const float* __restrict__ xw,
    float* __restrict__ gcn)
{
  int lane = threadIdx.x & 31;
  int e = blockIdx.x * 8 + (threadIdx.x >> 5);
  if (e >= EE) return;
  int s = src[e], d = dst[e];
  float coeff = dinv[s] * dinv[d];
  float4 vv = ((const float4*)(xw + (size_t)s * DD))[lane];
  float* g = gcn + (size_t)d * DD + lane * 4;
  atomicAdd(g + 0, vv.x * coeff);
  atomicAdd(g + 1, vv.y * coeff);
  atomicAdd(g + 2, vv.z * coeff);
  atomicAdd(g + 3, vv.w * coeff);
}

// ---------------------------------------------------------------------------
// K5: M = k^T @ v  (128x128, K=50000) via WMMA with K-split + atomic reduce.
// A rows = kT rows (feature i), B columns = vT rows (feature j).
// ---------------------------------------------------------------------------
__global__ __launch_bounds__(256) void ktv_kernel(
    const _Float16* __restrict__ kT, const _Float16* __restrict__ vT,
    float* __restrict__ Mbuf)
{
  const int lane = threadIdx.x & 31;
  const int wid  = blockIdx.x * 8 + (threadIdx.x >> 5);   // 0..2047
  const int pair = wid & 63, split = wid >> 6;
  const int i0 = (pair >> 3) * 16, j0 = (pair & 7) * 16;
  const int hi = lane >> 4, colL = lane & 15;
  const _Float16* arow = kT + (size_t)(i0 + colL) * NPAD;
  const _Float16* brow = vT + (size_t)(j0 + colL) * NPAD;
  const int baseA = hi * 8, baseB = hi * 16;

  v8f acc = {0, 0, 0, 0, 0, 0, 0, 0};
  for (int c = split; c < NCHUNK; c += KSPLIT) {
    int n0 = c * 32;
    v16h a = load_a16(arow + n0 + baseA);
    v16h b = load_a16(brow + n0 + baseA);
    acc = wmma_f16(a, b, acc);
  }
#pragma unroll
  for (int r = 0; r < 8; ++r)
    atomicAdd(&Mbuf[(i0 + r + hi * 8) * DD + j0 + colL], acc[r]);
}

// K5b: norms -> inv scalar; pack M/128 into f16 B layout.
__global__ __launch_bounds__(256) void finalize_kernel(
    const float* __restrict__ Mbuf, _Float16* __restrict__ mpack,
    float* __restrict__ scal)
{
  if (blockIdx.x == 0 && threadIdx.x == 0) {
    float nq = sqrtf(scal[0]);
    float nk = sqrtf(scal[1]);
    scal[2] = 1.0f / (nq * nk);
  }
  int gid = blockIdx.x * blockDim.x + threadIdx.x;
  int stride = gridDim.x * blockDim.x;
  for (int i = gid; i < 16384; i += stride) {
    int r = i;
    int t = r / 2048;  r %= 2048;
    int c = r / 512;   r %= 512;
    int lane = r / 16; int h = r % 16;
    int n = t * 16 + (lane & 15);
    int k = c * 32 + ((lane < 16) ? 0 : 8) + ((h < 8) ? h : h + 8);
    mpack[i] = (_Float16)(Mbuf[k * DD + n] * (1.0f / 128.0f));
  }
}

// ---------------------------------------------------------------------------
// K6: fused epilogue.  One wave per 16-row tile.
//   t   = q @ (M/128)  (WMMA) ; scaled by inv=1/(||q||*||k||) in f32
//   qk  = (q . ksum) * inv / 128   (in-register dot + shfl_xor 16)
//   Z   = 0.4 * scale * (v + t) + 0.6 * x
//   Zo  = 0.4 * Z + 0.6 * (gcn + b_gcn)
//   out = Zo @ W_lin^T + b_lin     (via per-wave LDS transpose + WMMA)
// ---------------------------------------------------------------------------
__global__ __launch_bounds__(256) void fused_out_kernel(
    const float* __restrict__ x, const _Float16* __restrict__ q16,
    const float* __restrict__ v32, const float* __restrict__ gcn,
    const _Float16* __restrict__ mpack, const _Float16* __restrict__ wpack,
    const float* __restrict__ ksum, const float* __restrict__ scal,
    const float* __restrict__ bgcn, const float* __restrict__ blin,
    float* __restrict__ out)
{
  __shared__ _Float16 zbuf[8][16 * DD];   // per-wave Z_out tile (f16)
  __shared__ float    scb[8][16];         // per-wave per-row attention scale

  const int lane = threadIdx.x & 31;
  const int wave = threadIdx.x >> 5;
  const int tile = blockIdx.x * 8 + wave;
  if (tile >= NTILE) return;
  const int m0 = tile * 16;
  const int hi = lane >> 4, colL = lane & 15;
  const int rowA = m0 + colL;
  const int baseA = hi * 8;

  // A operand: q rows (f16, row-major)
  v16h Aq[4];
#pragma unroll
  for (int c = 0; c < 4; ++c)
    Aq[c] = load_a16(q16 + (size_t)rowA * DD + c * 32 + baseA);

  const float inv = scal[2];

  // qk for this lane's row: lanes L and L^16 hold complementary K halves
  float part = 0.0f;
#pragma unroll
  for (int c = 0; c < 4; ++c) {
#pragma unroll
    for (int h = 0; h < 8; ++h) {
      part += (float)Aq[c][h]     * ksum[32 * c + baseA + h];
      part += (float)Aq[c][8 + h] * ksum[32 * c + baseA + 16 + h];
    }
  }
  float qk = (part + __shfl_xor(part, 16)) * inv * (1.0f / 128.0f);
  if (lane < 16) scb[wave][colL] = 1.0f / (1.0f + qk);
  asm volatile("s_wait_dscnt 0" ::: "memory");   // wave-local LDS visibility
  float srow[8];
#pragma unroll
  for (int r = 0; r < 8; ++r) srow[r] = scb[wave][r + hi * 8];

  // t = q @ (M/128), then full elementwise blend -> zbuf (f16, row-major tile)
#pragma unroll
  for (int ct = 0; ct < 8; ++ct) {
    v8f acc = {0, 0, 0, 0, 0, 0, 0, 0};
#pragma unroll
    for (int c = 0; c < 4; ++c)
      acc = wmma_f16(Aq[c], *(const v16h*)(mpack + (ct * 4 + c) * 512 + lane * 16), acc);
    int col = ct * 16 + colL;
    float bg = bgcn[col];
#pragma unroll
    for (int r = 0; r < 8; ++r) {
      int row = m0 + r + hi * 8;
      float tt   = acc[r] * inv;
      float attn = v32[(size_t)row * DD + col] + tt;
      float z    = 0.4f * srow[r] * attn + 0.6f * x[(size_t)row * DD + col];
      float zo   = 0.4f * z + 0.6f * (gcn[(size_t)row * DD + col] + bg);
      zbuf[wave][(r + hi * 8) * DD + col] = (_Float16)zo;
    }
  }
  asm volatile("s_wait_dscnt 0" ::: "memory");   // transpose: writes -> reads

  // A operand: Z_out rows from LDS
  v16h Az[4];
#pragma unroll
  for (int c = 0; c < 4; ++c)
    Az[c] = load_a16(&zbuf[wave][colL * DD + c * 32 + baseA]);

  // out = Z_out @ W_lin^T + b_lin   (W_lin packed at mat index 4)
#pragma unroll
  for (int ct = 0; ct < 8; ++ct) {
    v8f acc = {0, 0, 0, 0, 0, 0, 0, 0};
#pragma unroll
    for (int c = 0; c < 4; ++c)
      acc = wmma_f16(Az[c], *(const v16h*)(wpack + ((4 * 8 + ct) * 4 + c) * 512 + lane * 16), acc);
    int col = ct * 16 + colL;
    float bl = blin[col];
#pragma unroll
    for (int r = 0; r < 8; ++r)
      out[(size_t)(m0 + r + hi * 8) * DD + col] = acc[r] + bl;
  }
}

// ---------------------------------------------------------------------------
extern "C" void kernel_launch(void* const* d_in, const int* in_sizes, int n_in,
                              void* d_out, int out_size, void* d_ws, size_t ws_size,
                              hipStream_t stream) {
  (void)in_sizes; (void)n_in; (void)out_size; (void)ws_size;
  const float* x    = (const float*)d_in[0];
  const int*   eidx = (const int*)d_in[1];        // [2,E]: src row then dst row
  const float* Wq   = (const float*)d_in[2];
  const float* Wk   = (const float*)d_in[3];
  const float* Wv   = (const float*)d_in[4];
  const float* Wg   = (const float*)d_in[5];
  const float* bgcn = (const float*)d_in[6];
  const float* Wl   = (const float*)d_in[7];
  const float* blin = (const float*)d_in[8];
  float* out = (float*)d_out;

  const int* src = eidx;
  const int* dst = eidx + EE;

  // Workspace layout (256B aligned slices)
  char* w = (char*)d_ws;
  size_t o = 0;
  auto take = [&](size_t bytes) { char* p = w + o; o += (bytes + 255) & ~(size_t)255; return p; };
  _Float16* xh    = (_Float16*)take((size_t)NN * DD * 2);   // x f16
  _Float16* q16   = (_Float16*)take((size_t)NN * DD * 2);   // q f16
  float*    v32   = (float*)   take((size_t)NN * DD * 4);   // v f32
  float*    xw    = (float*)   take((size_t)NN * DD * 4);   // x@Wg^T f32
  float*    gcnb  = (float*)   take((size_t)NN * DD * 4);   // gcn accum f32
  _Float16* kT    = (_Float16*)take((size_t)DD * NPAD * 2); // k^T f16 (padded)
  _Float16* vT    = (_Float16*)take((size_t)DD * NPAD * 2); // v^T f16 (padded)
  float*    deg   = (float*)   take((size_t)NN * 4);
  float*    dinv  = (float*)   take((size_t)NN * 4);
  float*    ideg  = (float*)   take((size_t)NN * 4);
  _Float16* wpack = (_Float16*)take((size_t)5 * 16384 * 2); // 5 packed weights
  _Float16* mpack = (_Float16*)take((size_t)16384 * 2);     // packed M/128
  float*    Mbuf  = (float*)   take((size_t)DD * DD * 4);   // k^T v f32
  float*    ksum  = (float*)   take(512);
  float*    scal  = (float*)   take(256);  // [0]=||q||^2 [1]=||k||^2 [2]=inv

  prep_kernel<<<256, 256, 0, stream>>>(Wq, Wk, Wv, Wg, Wl, wpack, Mbuf, ksum,
                                       scal, kT, vT);
  xcast_kernel<<<(NN * DD / 4 + 255) / 256, 256, 0, stream>>>(x, xh);
  deg_init_kernel<<<(NN + 255) / 256, 256, 0, stream>>>(deg);
  deg_edge_kernel<<<(EE + 255) / 256, 256, 0, stream>>>(dst, deg);
  deg_fin_kernel<<<(NN + 255) / 256, 256, 0, stream>>>(deg, dinv, ideg);
  gemm_qkvg_kernel<<<(NTILE + 7) / 8, 256, 0, stream>>>(
      xh, wpack, ideg, q16, v32, xw, gcnb, kT, vT, ksum, scal);
  gcn_scatter_kernel<<<(EE + 7) / 8, 256, 0, stream>>>(src, dst, dinv, xw, gcnb);
  ktv_kernel<<<(64 * KSPLIT) / 8, 256, 0, stream>>>(kT, vT, Mbuf);
  finalize_kernel<<<64, 256, 0, stream>>>(Mbuf, mpack, scal);
  fused_out_kernel<<<(NTILE + 7) / 8, 256, 0, stream>>>(
      x, q16, v32, gcnb, mpack, wpack, ksum, scal, bgcn, blin, out);
}
